// AttentionLinks_21328807591969
// MI455X (gfx1250) — matrix operand order repaired and checked by reading, not compiled, not run on hardware
//
#include <hip/hip_runtime.h>
#include <hip/hip_bf16.h>
#include <math.h>

#define BATCH 4
#define LSEQ  2048
#define EMBD  512
#define HID   64
#define NEGV  -1e9f

typedef __attribute__((ext_vector_type(16))) _Float16 v16h;
typedef __attribute__((ext_vector_type(8)))  _Float16 v8h;
typedef __attribute__((ext_vector_type(8)))  float    v8f;

#define WMMA(A, B, C) __builtin_amdgcn_wmma_f32_16x16x32_f16( \
    false, (A), false, (B), (short)0, (C), false, false)

// ---------------------------------------------------------------------------
// WMMA fragment helpers (CDNA5 ISA 7.12.2, wave32) — vectorized b128 loads.
// Requires K-contiguous storage: element(m,k) = base[m*ld + k].
// A 16x32 f16: lane<16 -> elems 0..7 = K0..7, 8..15 = K16..23; lane>=16 -> +8.
// ---------------------------------------------------------------------------
__device__ inline v16h frag_a_rowk(const _Float16* base, int ld, int lane) {
  int m = lane & 15, hi = (lane >> 4) & 1;
  const _Float16* p = base + m * ld;
  v8h a0 = *(const v8h*)(p + (hi ? 8 : 0));
  v8h a1 = *(const v8h*)(p + (hi ? 24 : 16));
  return __builtin_shufflevector(a0, a1, 0, 1, 2, 3, 4, 5, 6, 7,
                                 8, 9, 10, 11, 12, 13, 14, 15);
}
// B 32x16 f16: lane<16 -> N=lane, elems 0..7 = K0..7, 8..15 = K16..23; lane>=16 -> N=lane-16, +8.
// element(k,n) = base[n*ld + k]  (n-major, k-contiguous)
__device__ inline v16h frag_b_rowk(const _Float16* base, int ld, int lane) {
  int n = lane & 15, hi = (lane >> 4) & 1;
  const _Float16* p = base + n * ld + (hi ? 8 : 0);
  v8h b0 = *(const v8h*)(p);
  v8h b1 = *(const v8h*)(p + 16);
  return __builtin_shufflevector(b0, b1, 0, 1, 2, 3, 4, 5, 6, 7,
                                 8, 9, 10, 11, 12, 13, 14, 15);
}
// C/D 16x16 f32: VGPR v -> M = v + hi*8, N = lane%16
__device__ inline void frag_store_f32(float* base, size_t ldc, int lane, v8f d) {
  int n = lane & 15, hi = (lane >> 4) & 1;
#pragma unroll
  for (int v = 0; v < 8; ++v) base[(size_t)(v + hi * 8) * ldc + n] = d[v];
}
__device__ inline void frag_store_f16(_Float16* base, int ldc, int lane, v8f d) {
  int n = lane & 15, hi = (lane >> 4) & 1;
#pragma unroll
  for (int v = 0; v < 8; ++v) base[(v + hi * 8) * ldc + n] = (_Float16)d[v];
}

// ---------------------------------------------------------------------------
// Kernel 1: WMMA projection  qraw = x@Wq^T, kraw = x@Wk^T  (M=8192,N=64,K=512)
// fused with layernorm + l2norm; emits f16 q/k.  One block per 128 rows.
// ---------------------------------------------------------------------------
__global__ __launch_bounds__(256) void projgemm_kernel(
    const float* __restrict__ x, const float* __restrict__ Wq,
    const float* __restrict__ Wk,
    const float* __restrict__ gq, const float* __restrict__ bq,
    const float* __restrict__ gk, const float* __restrict__ bk,
    _Float16* __restrict__ qh, _Float16* __restrict__ kh) {
  __shared__ __attribute__((aligned(16))) _Float16 xs[128 * 64];   // 16 KB
  __shared__ __attribute__((aligned(16))) _Float16 wqs[64 * 64];   //  8 KB
  __shared__ __attribute__((aligned(16))) _Float16 wks[64 * 64];   //  8 KB
  __shared__ __attribute__((aligned(16))) float    raw[128 * 64];  // 32 KB
  int tid = threadIdx.x;
  int lane = tid & 31, wave = tid >> 5;
  size_t row0 = (size_t)blockIdx.x * 128;

  v8f accq[4] = {}, acck[4] = {};
  for (int c = 0; c < EMBD / 64; ++c) {
    __syncthreads();
    // stage x chunk (128x64) as f16
    for (int i = tid; i < 2048; i += 256) {       // 2048 float4 = 8192 vals
      int r = i >> 4, off = (i & 15) * 4;
      float4 v = *(const float4*)(x + (row0 + r) * EMBD + c * 64 + off);
      _Float16* d = xs + r * 64 + off;
      d[0] = (_Float16)v.x; d[1] = (_Float16)v.y;
      d[2] = (_Float16)v.z; d[3] = (_Float16)v.w;
    }
    // stage W chunks (64x64) as f16 (n-major rows, k contiguous)
    for (int i = tid; i < 1024; i += 256) {
      int h = i >> 4, off = (i & 15) * 4;
      float4 vq = *(const float4*)(Wq + (size_t)h * EMBD + c * 64 + off);
      float4 vk = *(const float4*)(Wk + (size_t)h * EMBD + c * 64 + off);
      _Float16* dq = wqs + h * 64 + off;
      _Float16* dk = wks + h * 64 + off;
      dq[0] = (_Float16)vq.x; dq[1] = (_Float16)vq.y;
      dq[2] = (_Float16)vq.z; dq[3] = (_Float16)vq.w;
      dk[0] = (_Float16)vk.x; dk[1] = (_Float16)vk.y;
      dk[2] = (_Float16)vk.z; dk[3] = (_Float16)vk.w;
    }
    __syncthreads();
    v16h a0 = frag_a_rowk(xs + wave * 16 * 64, 64, lane);
    v16h a1 = frag_a_rowk(xs + wave * 16 * 64 + 32, 64, lane);
#pragma unroll
    for (int tn = 0; tn < 4; ++tn) {
      v16h b0 = frag_b_rowk(wqs + tn * 16 * 64, 64, lane);
      v16h b1 = frag_b_rowk(wqs + tn * 16 * 64 + 32, 64, lane);
      accq[tn] = WMMA(a0, b0, accq[tn]);
      accq[tn] = WMMA(a1, b1, accq[tn]);
      b0 = frag_b_rowk(wks + tn * 16 * 64, 64, lane);
      b1 = frag_b_rowk(wks + tn * 16 * 64 + 32, 64, lane);
      acck[tn] = WMMA(a0, b0, acck[tn]);
      acck[tn] = WMMA(a1, b1, acck[tn]);
    }
  }

  // ---- q: LN + l2norm ----
  __syncthreads();
#pragma unroll
  for (int tn = 0; tn < 4; ++tn)
    frag_store_f32(raw + wave * 16 * 64 + tn * 16, 64, lane, accq[tn]);
  __syncthreads();
  if (tid < 128) {
    const float* r = raw + tid * 64;
    float s = 0.f;
    for (int j = 0; j < 64; ++j) s += r[j];
    float mu = s * (1.f / 64.f), vv = 0.f;
    for (int j = 0; j < 64; ++j) { float d = r[j] - mu; vv += d * d; }
    float inv = rsqrtf(vv * (1.f / 64.f) + 1e-5f);
    float nn = 0.f;
    float ln[64];
    for (int j = 0; j < 64; ++j) { ln[j] = (r[j] - mu) * inv * gq[j] + bq[j]; nn += ln[j] * ln[j]; }
    float sc = 1.f / fmaxf(sqrtf(nn), 1e-6f);
    _Float16* o = qh + (row0 + tid) * HID;
    for (int j = 0; j < 64; ++j) o[j] = (_Float16)(ln[j] * sc);
  }
  // ---- k: LN + l2norm ----
  __syncthreads();
#pragma unroll
  for (int tn = 0; tn < 4; ++tn)
    frag_store_f32(raw + wave * 16 * 64 + tn * 16, 64, lane, acck[tn]);
  __syncthreads();
  if (tid < 128) {
    const float* r = raw + tid * 64;
    float s = 0.f;
    for (int j = 0; j < 64; ++j) s += r[j];
    float mu = s * (1.f / 64.f), vv = 0.f;
    for (int j = 0; j < 64; ++j) { float d = r[j] - mu; vv += d * d; }
    float inv = rsqrtf(vv * (1.f / 64.f) + 1e-5f);
    float nn = 0.f;
    float ln[64];
    for (int j = 0; j < 64; ++j) { ln[j] = (r[j] - mu) * inv * gk[j] + bk[j]; nn += ln[j] * ln[j]; }
    float sc = 1.f / fmaxf(sqrtf(nn), 1e-6f);
    _Float16* o = kh + (row0 + tid) * HID;
    for (int j = 0; j < 64; ++j) o[j] = (_Float16)(ln[j] * sc);
  }
}

// ---------------------------------------------------------------------------
// Kernel 2: Gram  Gkk = k^T k, Gkq = k^T q  [B,64,64], K-dim = L = 2048.
// Chunks staged TRANSPOSED in LDS so fragments are k-contiguous b128 loads.
// ---------------------------------------------------------------------------
__global__ __launch_bounds__(512) void gram_kernel(
    const _Float16* __restrict__ qh, const _Float16* __restrict__ kh,
    _Float16* __restrict__ Gkk, _Float16* __restrict__ Gkq) {
  __shared__ __attribute__((aligned(16))) _Float16 lkT[HID * 32];  // [d][l]
  __shared__ __attribute__((aligned(16))) _Float16 lxT[HID * 32];
  int tid = threadIdx.x;
  int which = blockIdx.x, b = blockIdx.y;
  const _Float16* xsrc = which ? qh : kh;
  int lane = tid & 31, wave = tid >> 5;
  int tm = wave >> 2, tn = wave & 3;
  v8f acc = {};
  for (int c = 0; c < LSEQ / 32; ++c) {
    __syncthreads();
    {   // 512 threads: 0..255 stage lkT, 256..511 stage lxT (v8h reads)
      int j = tid & 255;
      int lloc = j >> 3, d0 = (j & 7) * 8;
      const _Float16* src = (tid < 256) ? kh : xsrc;
      _Float16* dst = (tid < 256) ? lkT : lxT;
      v8h v = *(const v8h*)(src + ((size_t)b * LSEQ + c * 32 + lloc) * HID + d0);
#pragma unroll
      for (int e = 0; e < 8; ++e) dst[(d0 + e) * 32 + lloc] = v[e];
    }
    if (c + 1 < LSEQ / 32)
      __builtin_prefetch(kh + ((size_t)b * LSEQ + (c + 1) * 32) * HID, 0, 1);
    __syncthreads();
    // A(m,k)=k[l=k][d=tm*16+m]=lkT[(tm*16+m)*32+k]; B(k,n)=lxT[(tn*16+n)*32+k]
    v16h a  = frag_a_rowk(lkT + tm * 16 * 32, 32, lane);
    v16h bf = frag_b_rowk(lxT + tn * 16 * 32, 32, lane);
    acc = WMMA(a, bf, acc);
  }
  _Float16* G = which ? Gkq : Gkk;
  frag_store_f16(G + (size_t)b * HID * HID + tm * 16 * HID + tn * 16,
                 HID, lane, acc);
}

// ---------------------------------------------------------------------------
// Kernel 3: Aq = q @ Gkk, Af = q @ Gkq   [B,L,64] f16.  M=2048,N=64,K=64.
// G staged transposed (n-major) for contiguous B fragments.
// ---------------------------------------------------------------------------
__global__ __launch_bounds__(256) void smallgemm_kernel(
    const _Float16* __restrict__ qh,
    const _Float16* __restrict__ Gkk, const _Float16* __restrict__ Gkq,
    _Float16* __restrict__ Aq, _Float16* __restrict__ Af) {
  __shared__ __attribute__((aligned(16))) _Float16 qt[64 * HID];
  __shared__ __attribute__((aligned(16))) _Float16 GsT[2][HID * HID];
  int tid = threadIdx.x;
  int l0 = blockIdx.x * 64, b = blockIdx.y;
  for (int i = tid; i < 512; i += 256) {         // qt: 512 v8h
    int r = i >> 3, off = (i & 7) * 8;
    *(v8h*)(qt + r * HID + off) =
        *(const v8h*)(qh + ((size_t)b * LSEQ + l0 + r) * HID + off);
  }
  for (int t = tid; t < HID * HID; t += 256) {   // transpose G into n-major
    GsT[0][(t & 63) * HID + (t >> 6)] = Gkk[(size_t)b * HID * HID + t];
    GsT[1][(t & 63) * HID + (t >> 6)] = Gkq[(size_t)b * HID * HID + t];
  }
  __syncthreads();
  int lane = tid & 31, wave = tid >> 5;
  for (int job = wave; job < 32; job += 8) {
    int out = job >> 4, t = job & 15;
    int tm = t >> 2, tn = t & 3;
    v8f acc = {};
#pragma unroll
    for (int kb = 0; kb < HID; kb += 32) {
      v16h a  = frag_a_rowk(qt + tm * 16 * HID + kb, HID, lane);
      v16h bf = frag_b_rowk(GsT[out] + tn * 16 * HID + kb, HID, lane);
      acc = WMMA(a, bf, acc);
    }
    _Float16* O = out ? Af : Aq;
    frag_store_f16(O + ((size_t)b * LSEQ + l0 + tm * 16) * HID + tn * 16,
                   HID, lane, acc);
  }
}

// ---------------------------------------------------------------------------
// Kernel 4: S = A @ X^T per batch; M=N=2048, K=64.  128x128 tile per block;
// each wave owns a 16-row strip and reuses its A fragments across 8 B tiles.
// ---------------------------------------------------------------------------
__global__ __launch_bounds__(256) void biggemm_kernel(
    const _Float16* __restrict__ A, const _Float16* __restrict__ X,
    float* __restrict__ S) {
  __shared__ __attribute__((aligned(16))) _Float16 At[128 * HID]; // 16 KB
  __shared__ __attribute__((aligned(16))) _Float16 Xt[128 * HID]; // 16 KB
  int tid = threadIdx.x;
  int m0 = blockIdx.x * 128, n0 = blockIdx.y * 128, b = blockIdx.z;
  for (int i = tid; i < 1024; i += 256) {        // 1024 v8h per array
    int r = i >> 3, off = (i & 7) * 8;
    *(v8h*)(At + r * HID + off) =
        *(const v8h*)(A + ((size_t)b * LSEQ + m0 + r) * HID + off);
    *(v8h*)(Xt + r * HID + off) =
        *(const v8h*)(X + ((size_t)b * LSEQ + n0 + r) * HID + off);
  }
  __syncthreads();
  int lane = tid & 31, wave = tid >> 5;
  v16h a0 = frag_a_rowk(At + wave * 16 * HID, HID, lane);
  v16h a1 = frag_a_rowk(At + wave * 16 * HID + 32, HID, lane);
  float* srow = S + (size_t)b * LSEQ * LSEQ + (size_t)(m0 + wave * 16) * LSEQ + n0;
#pragma unroll
  for (int tn = 0; tn < 8; ++tn) {
    v16h b0 = frag_b_rowk(Xt + tn * 16 * HID, HID, lane);
    v16h b1 = frag_b_rowk(Xt + tn * 16 * HID + 32, HID, lane);
    v8f acc = {};
    acc = WMMA(a0, b0, acc);
    acc = WMMA(a1, b1, acc);
    frag_store_f32(srow + tn * 16, LSEQ, lane, acc);
  }
}

// ---------------------------------------------------------------------------
// entmax-1.5 over rows of 2048: bitonic sort + block scans in LDS.
// ---------------------------------------------------------------------------
__device__ inline float reduce_val_sum(float v, float* red) {
  int tid = threadIdx.x;
  red[tid] = v; __syncthreads();
  for (int s = 128; s > 0; s >>= 1) { if (tid < s) red[tid] += red[tid + s]; __syncthreads(); }
  float r = red[0]; __syncthreads(); return r;
}
__device__ inline float reduce_val_max(float v, float* red) {
  int tid = threadIdx.x;
  red[tid] = v; __syncthreads();
  for (int s = 128; s > 0; s >>= 1) { if (tid < s) red[tid] = fmaxf(red[tid], red[tid + s]); __syncthreads(); }
  float r = red[0]; __syncthreads(); return r;
}
__device__ inline void scan_inplace(float* a, float* red) {
  int tid = threadIdx.x;
  int base = tid * (LSEQ / 256);
  float s = 0.f;
#pragma unroll
  for (int i = 0; i < LSEQ / 256; ++i) { s += a[base + i]; a[base + i] = s; }
  red[tid] = s; __syncthreads();
  if (tid == 0) { float t = 0.f; for (int i = 0; i < 256; ++i) { t += red[i]; red[i] = t; } }
  __syncthreads();
  float off = tid ? red[tid - 1] : 0.f;
#pragma unroll
  for (int i = 0; i < LSEQ / 256; ++i) a[base + i] += off;
  __syncthreads();
}
__device__ inline void bitonic_desc(float* a) {
  int tid = threadIdx.x;
  for (int k = 2; k <= LSEQ; k <<= 1) {
    for (int j = k >> 1; j > 0; j >>= 1) {
      for (int t = tid; t < LSEQ; t += 256) {
        int ixj = t ^ j;
        if (ixj > t) {
          bool dir = ((t & k) == 0);
          float av = a[t], bv = a[ixj];
          bool sw = dir ? (av < bv) : (av > bv);
          if (sw) { a[t] = bv; a[ixj] = av; }
        }
      }
      __syncthreads();
    }
  }
}

__global__ __launch_bounds__(256) void entmax_kernel(
    const float* __restrict__ S, float* __restrict__ P,
    const float* __restrict__ wp, int do_center, int do_clip) {
  __shared__ float z[LSEQ];
  __shared__ float zs[LSEQ];
  __shared__ float cs[LSEQ];
  __shared__ float cq[LSEQ];
  __shared__ float red[256];
  int tid = threadIdx.x;
  size_t row = blockIdx.x;
  const float* srow = S + row * LSEQ;
  float w = 1.f;
  if (wp) w = 2.f / (1.f + expf(-wp[0]));

  for (int i = tid; i < LSEQ / 4; i += 256)
    ((float4*)z)[i] = ((const float4*)srow)[i];
  __syncthreads();
  float mean = 0.f;
  if (do_center) {
    float s = 0.f;
    for (int i = tid; i < LSEQ; i += 256) s += z[i];
    mean = reduce_val_sum(s, red) * (1.f / LSEQ);
  }
  for (int i = tid; i < LSEQ; i += 256) z[i] = 0.5f * w * (z[i] - mean);
  __syncthreads();
  float m = -3.4e38f;
  for (int i = tid; i < LSEQ; i += 256) m = fmaxf(m, z[i]);
  float mx = reduce_val_max(m, red);
  for (int i = tid; i < LSEQ; i += 256) { float v = z[i] - mx; z[i] = v; zs[i] = v; }
  __syncthreads();
  bitonic_desc(zs);
  for (int i = tid; i < LSEQ; i += 256) { float v = zs[i]; cs[i] = v; cq[i] = v * v; }
  __syncthreads();
  scan_inplace(cs, red);
  scan_inplace(cq, red);
  float flags = 0.f;
  for (int i = tid; i < LSEQ; i += 256) {
    float kk    = (float)(i + 1);
    float mk    = cs[i] / kk;
    float ss    = cq[i] - cs[i] * mk;
    float delta = (1.f - ss) / kk;
    float tau   = mk - sqrtf(fmaxf(delta, 0.f));
    flags += (tau <= zs[i]) ? 1.f : 0.f;
    cs[i] = tau;
  }
  __syncthreads();
  float fsum = reduce_val_sum(flags, red);
  int support = (int)(fsum + 0.5f);
  if (support < 1) support = 1;
  float tau_star = cs[support - 1];
  float clipv = do_clip ? (1.f - 1e-6f) : 3.4e38f;
  float* prow = P + row * LSEQ;
  for (int i = tid; i < LSEQ / 4; i += 256) {
    float4 o;
    float p0 = fmaxf(z[i * 4 + 0] - tau_star, 0.f);
    float p1 = fmaxf(z[i * 4 + 1] - tau_star, 0.f);
    float p2 = fmaxf(z[i * 4 + 2] - tau_star, 0.f);
    float p3 = fmaxf(z[i * 4 + 3] - tau_star, 0.f);
    o.x = fminf(p0 * p0, clipv); o.y = fminf(p1 * p1, clipv);
    o.z = fminf(p2 * p2, clipv); o.w = fminf(p3 * p3, clipv);
    ((float4*)prow)[i] = o;
  }
}

// ---------------------------------------------------------------------------
// Dehubbing: rsqrt(col_sum(pC)+eps)
// ---------------------------------------------------------------------------
__global__ __launch_bounds__(256) void colfac_kernel(
    const float* __restrict__ pC, float* __restrict__ fac) {
  int idx = blockIdx.x * 256 + threadIdx.x;
  int b = idx / LSEQ, mcol = idx % LSEQ;
  const float* base = pC + (size_t)b * LSEQ * LSEQ + mcol;
  float s = 0.f;
  for (int l = 0; l < LSEQ; ++l) {
    if ((l & 7) == 0 && l + 8 < LSEQ)
      __builtin_prefetch(base + (size_t)(l + 8) * LSEQ, 0, 1);
    s += base[(size_t)l * LSEQ];
  }
  fac[idx] = rsqrtf(s + 1e-6f);
}

// ---------------------------------------------------------------------------
// Scale pC in place + harmonic fusion + diag mask -> H_pre
// ---------------------------------------------------------------------------
__global__ __launch_bounds__(256) void fuse_kernel(
    float* __restrict__ pC, const float* __restrict__ pF,
    const float* __restrict__ fac, float* __restrict__ H) {
  size_t i = (size_t)blockIdx.x * 256 + threadIdx.x;
  size_t b = i / ((size_t)LSEQ * LSEQ);
  size_t rem = i % ((size_t)LSEQ * LSEQ);
  int r = (int)(rem / LSEQ), mcol = (int)(rem % LSEQ);
  float c = pC[i] * fac[b * LSEQ + mcol];
  pC[i] = c;
  float f = pF[i];
  float h = 2.f * c * f / (c + f + 1e-6f);
  if (r == mcol) h = NEGV;
  H[i] = h;
}

// ---------------------------------------------------------------------------
extern "C" void kernel_launch(void* const* d_in, const int* in_sizes, int n_in,
                              void* d_out, int out_size, void* d_ws, size_t ws_size,
                              hipStream_t stream) {
  (void)in_sizes; (void)n_in; (void)out_size; (void)ws_size;
  const float* x  = (const float*)d_in[0];
  const float* Wq = (const float*)d_in[1];
  const float* Wk = (const float*)d_in[2];
  const float* gq = (const float*)d_in[3];
  const float* bq = (const float*)d_in[4];
  const float* gk = (const float*)d_in[5];
  const float* bk = (const float*)d_in[6];
  const float* Cw = (const float*)d_in[7];
  const float* Fw = (const float*)d_in[8];

  float* out = (float*)d_out;
  const size_t LL = (size_t)LSEQ * LSEQ;
  float* Hreg  = out;                       // scratch for raw scores, final H
  float* pCreg = out + (size_t)BATCH * LL;
  float* pFreg = out + 2 * (size_t)BATCH * LL;

  char* ws = (char*)d_ws;
  _Float16* qh  = (_Float16*)ws; ws += (size_t)BATCH * LSEQ * HID * 2;
  _Float16* kh  = (_Float16*)ws; ws += (size_t)BATCH * LSEQ * HID * 2;
  _Float16* Gkk = (_Float16*)ws; ws += (size_t)BATCH * HID * HID * 2;
  _Float16* Gkq = (_Float16*)ws; ws += (size_t)BATCH * HID * HID * 2;
  _Float16* Aq  = (_Float16*)ws; ws += (size_t)BATCH * LSEQ * HID * 2;
  _Float16* Af  = (_Float16*)ws; ws += (size_t)BATCH * LSEQ * HID * 2;
  float*    fac = (float*)ws;    ws += (size_t)BATCH * LSEQ * 4;

  projgemm_kernel<<<BATCH * LSEQ / 128, 256, 0, stream>>>(
      x, Wq, Wk, gq, bq, gk, bk, qh, kh);
  gram_kernel<<<dim3(2, BATCH), 512, 0, stream>>>(qh, kh, Gkk, Gkq);
  smallgemm_kernel<<<dim3(LSEQ / 64, BATCH), 256, 0, stream>>>(
      qh, Gkk, Gkq, Aq, Af);

  // C path: C_raw -> entmax -> pC
  biggemm_kernel<<<dim3(LSEQ / 128, LSEQ / 128, BATCH), 256, 0, stream>>>(Aq, qh, Hreg);
  entmax_kernel<<<BATCH * LSEQ, 256, 0, stream>>>(Hreg, pCreg, Cw, 1, 1);
  // F path: F_raw -> entmax -> pF
  biggemm_kernel<<<dim3(LSEQ / 128, LSEQ / 128, BATCH), 256, 0, stream>>>(Af, kh, Hreg);
  entmax_kernel<<<BATCH * LSEQ, 256, 0, stream>>>(Hreg, pFreg, Fw, 1, 1);
  // dehub + harmonic fusion + final entmax
  colfac_kernel<<<(BATCH * LSEQ) / 256, 256, 0, stream>>>(pCreg, fac);
  fuse_kernel<<<(unsigned)((BATCH * LL) / 256), 256, 0, stream>>>(pCreg, pFreg, fac, Hreg);
  entmax_kernel<<<BATCH * LSEQ, 256, 0, stream>>>(Hreg, Hreg, nullptr, 0, 0);
}
